// MHN_WITH_1_HIDDEN_LAYER_2147483648712
// MI455X (gfx1250) — matrix operand order repaired and checked by reading, not compiled, not run on hardware
//
#include <hip/hip_runtime.h>

// Modern Hopfield update, factorized:
//   S = v * K^T            (GEMM1, M=B=1024, N=N2=1024, K=N1=256)  -- direct-global WMMA
//   W = exp(BETA*S - BETA/2*||k_n2||^2)   (exp(-BETA/2||v_b||^2) cancels in num/den)
//   den[b] = sum_n2 W[b,n2]
//   M = W * K              (GEMM2, M=B=1024, N=N1=256, K=N2=1024)  -- TDM->LDS WMMA
//   out = v + ALPHA * (M/den - v) * mask

#define BETA  0.01f
#define ALPHA 0.1f
#define B_DIM  1024
#define N1_DIM 256
#define N2_DIM 1024

typedef __attribute__((ext_vector_type(16))) _Float16 v16h;
typedef __attribute__((ext_vector_type(8)))  _Float16 v8h;
typedef __attribute__((ext_vector_type(8)))  float    v8f;
typedef __attribute__((ext_vector_type(4)))  unsigned int v4u;
typedef __attribute__((ext_vector_type(8)))  int          v8i;
typedef __attribute__((ext_vector_type(4)))  int          v4i;

// ---------------------------------------------------------------------------
// WMMA fragment helpers (layouts per CDNA5 ISA 7.12.2, wave32).
// A (16x32 f16 MxK): lane m = lane&15; lanes 0-15: K={+0..7,+16..23},
// lanes 16-31: K={+8..15,+24..31}. Two contiguous 16B chunks per lane.
// B (32x16 f16 KxN): lane n = lane&15; lanes 0-15: K={+0..7,+16..23},
// lanes 16-31: K={+8..15,+24..31} (same k order as A; K-contiguous storage).
// C/D (16x16 f32): lane n = lane&15; VGPR r holds row m = r + 8*(lane>>4).
// ---------------------------------------------------------------------------
__device__ __forceinline__ v16h frag_a_ptr(const _Float16* p /* row base + k0 */,
                                           int khalf) {
  v8h lo = *reinterpret_cast<const v8h*>(p + khalf * 8);
  v8h hi = *reinterpret_cast<const v8h*>(p + khalf * 8 + 16);
  v16h a;
#pragma unroll
  for (int i = 0; i < 8; ++i) { a[i] = lo[i]; a[i + 8] = hi[i]; }
  return a;
}

__device__ __forceinline__ v16h frag_b_ptr(const _Float16* p /* col base + k0 */,
                                           int khalf) {
  v8h lo = *reinterpret_cast<const v8h*>(p + khalf * 8);
  v8h hi = *reinterpret_cast<const v8h*>(p + khalf * 8 + 16);
  v16h b;
#pragma unroll
  for (int i = 0; i < 8; ++i) { b[i] = lo[i]; b[i + 8] = hi[i]; }
  return b;
}

// ---------------------------------------------------------------------------
// Tensor Data Mover: 2D tile (rows x cols of f16, X = contiguous dim) -> LDS.
// Descriptor packing per CDNA5 ISA ch.8.3/8.4 (D# group0: count/lds/global/type,
// group1: data_size, tensor dims, tile dims, dim0 stride). Groups 2/3 zero (2D).
// This toolchain exposes the 6-arg builtin: (v4u, v8i, v4i, v4i, v8i, cpol).
// ---------------------------------------------------------------------------
__device__ __forceinline__ unsigned lds_byte_off(const void* p) {
  // Flat LDS pointers carry the LDS offset in the low 32 bits (ISA 10.2).
  return (unsigned)(unsigned long long)(uintptr_t)p;
}

__device__ __forceinline__ void tdm_load_2d_f16(unsigned lds_off, const _Float16* gptr,
                                                unsigned tensor_d0, unsigned tensor_d1,
                                                unsigned tile_d0, unsigned tile_d1,
                                                unsigned d0_stride_elems) {
  unsigned long long ga = (unsigned long long)(uintptr_t)gptr;
  v4u g0;
  g0[0] = 1u;                                             // count=1, load, user
  g0[1] = lds_off;                                        // lds_addr (bytes)
  g0[2] = (unsigned)(ga & 0xFFFFFFFFu);                   // global_addr[31:0]
  g0[3] = (unsigned)((ga >> 32) & 0x01FFFFFFu) | (2u << 30); // addr[56:32] | type=2
  v8i g1;
  g1[0] = (int)(1u << 16);                                // data_size=1 (2 bytes)
  g1[1] = (int)((tensor_d0 & 0xFFFFu) << 16);             // tensor_dim0[15:0]
  g1[2] = (int)((tensor_d0 >> 16) | ((tensor_d1 & 0xFFFFu) << 16));
  g1[3] = (int)((tensor_d1 >> 16) | (tile_d0 << 16));     // tile_dim0
  g1[4] = (int)(tile_d1 & 0xFFFFu);                       // tile_dim1 (tile_dim2=0)
  g1[5] = (int)d0_stride_elems;                           // tensor_dim0_stride[31:0]
  g1[6] = 0;                                              // stride hi / dim1_stride lo
  g1[7] = 0;
  v4i z4 = {0, 0, 0, 0};
  v8i z8 = {0, 0, 0, 0, 0, 0, 0, 0};
  __builtin_amdgcn_tensor_load_to_lds(g0, g1, z4, z4, z8, 0);
}

// ---- prep: v -> f16 --------------------------------------------------------
__global__ void prep_v_kernel(const float* __restrict__ v, _Float16* __restrict__ vh) {
  int i = blockIdx.x * blockDim.x + threadIdx.x;
  vh[i] = (_Float16)v[i];
}

// ---- prep: kernel -> f16 (row-major + transposed) and ||k_n2||^2 ----------
__global__ void prep_k_kernel(const float* __restrict__ k, _Float16* __restrict__ kh,
                              _Float16* __restrict__ kth, float* __restrict__ kw) {
  __shared__ float s[N1_DIM];
  const int row = blockIdx.x;     // n2
  const int c   = threadIdx.x;    // n1
  float val = k[(size_t)row * N1_DIM + c];
  _Float16 h = (_Float16)val;
  kh[(size_t)row * N1_DIM + c]  = h;
  kth[(size_t)c * N2_DIM + row] = h;   // kT[n1][n2] -> K-contiguous B for GEMM2
  s[c] = val * val;
  __syncthreads();
#pragma unroll
  for (int off = N1_DIM / 2; off > 0; off >>= 1) {
    if (c < off) s[c] += s[c + off];
    __syncthreads();
  }
  if (c == 0) kw[row] = s[0];
}

// ---- GEMM1 + exp epilogue: W[b,n2] = exp(BETA*(v.k) - BETA/2*||k||^2) -----
// K=256 is small; L2-resident direct global loads, prefetch next K-slab.
__global__ void gemm1_w_kernel(const _Float16* __restrict__ vh,
                               const _Float16* __restrict__ kh,
                               const float* __restrict__ kw,
                               _Float16* __restrict__ wh) {
  const int wave = threadIdx.x >> 5;
  const int tile = blockIdx.x * 8 + wave;        // 64 x 64 tiles of 16x16
  const int bm = (tile >> 6) * 16;
  const int n0 = (tile & 63) * 16;
  const int lane  = threadIdx.x & 31;
  const int khalf = lane >> 4;
  const _Float16* arow = vh + (size_t)(bm + (lane & 15)) * N1_DIM;
  const _Float16* brow = kh + (size_t)(n0 + (lane & 15)) * N1_DIM;

  v8f c = {};
#pragma unroll
  for (int k0 = 0; k0 < N1_DIM; k0 += 32) {
    if (k0 + 32 < N1_DIM) {
      __builtin_prefetch((const void*)(arow + k0 + 32), 0, 0);
      __builtin_prefetch((const void*)(brow + k0 + 32), 0, 0);
    }
    v16h a = frag_a_ptr(arow + k0, khalf);
    v16h b = frag_b_ptr(brow + k0, khalf);
    c = __builtin_amdgcn_wmma_f32_16x16x32_f16(false, a, false, b,
                                               (short)0, c, false, false);
  }
  const int n2 = n0 + (lane & 15);
  const int mh = khalf * 8;
  const float bias = -0.5f * BETA * kw[n2];
#pragma unroll
  for (int r = 0; r < 8; ++r) {
    const int brow_i = bm + mh + r;
    float w = __expf(BETA * c[r] + bias);
    wh[(size_t)brow_i * N2_DIM + n2] = (_Float16)w;
  }
}

// ---- den[b] = sum_n2 W[b,n2] ----------------------------------------------
__global__ void den_kernel(const _Float16* __restrict__ wh, float* __restrict__ den) {
  __shared__ float s[256];
  const int b = blockIdx.x;
  const int t = threadIdx.x;
  float acc = 0.f;
#pragma unroll
  for (int j = t; j < N2_DIM; j += 256) acc += (float)wh[(size_t)b * N2_DIM + j];
  s[t] = acc;
  __syncthreads();
#pragma unroll
  for (int off = 128; off > 0; off >>= 1) {
    if (t < off) s[t] += s[t + off];
    __syncthreads();
  }
  if (t == 0) den[b] = s[0];
}

// ---- GEMM2 + fused Hopfield epilogue (TDM -> LDS double-buffered) ---------
// Block: 8 waves, covers 16 batch rows x 128 n1 columns. K=1024 in 16 slabs
// of 64. Wave 0 TDMs the shared A slab (W tile, reused by all 8 waves);
// each wave TDMs its private B slab (kT tile). s_wait_tensorcnt keeps the
// next slab's DMA in flight while WMMAs consume the current one from LDS.
#define KC 64
#define NCHUNK (N2_DIM / KC)

__global__ void gemm2_out_kernel(const _Float16* __restrict__ wh,
                                 const _Float16* __restrict__ kth,
                                 const float* __restrict__ den,
                                 const float* __restrict__ v,
                                 const float* __restrict__ mask,
                                 float* __restrict__ out) {
  __shared__ _Float16 sA[2][16 * KC];        // 4 KB  (shared W slab)
  __shared__ _Float16 sB[2][8][16 * KC];     // 32 KB (per-wave kT slabs)

  const int wave = threadIdx.x >> 5;
  const int lane = threadIdx.x & 31;
  const int khalf = lane >> 4;
  const int bm = (blockIdx.x >> 1) * 16;             // 64 b-tiles
  const int n0 = (blockIdx.x & 1) * 128 + wave * 16; // 2 column groups of 128

  const _Float16* gA = wh  + (size_t)bm * N2_DIM;    // A slab origin (row bm)
  const _Float16* gB = kth + (size_t)n0 * N2_DIM;    // B slab origin (col n0)

  // stage slab c into buffer buf
  auto stage = [&](int buf, int c) {
    const int k0 = c * KC;
    if (wave == 0)
      tdm_load_2d_f16(lds_byte_off(&sA[buf][0]), gA + k0,
                      N2_DIM, B_DIM, KC, 16, N2_DIM);
    tdm_load_2d_f16(lds_byte_off(&sB[buf][wave][0]), gB + k0,
                    N2_DIM, N1_DIM, KC, 16, N2_DIM);
  };

  stage(0, 0);

  v8f c = {};
  for (int ch = 0; ch < NCHUNK; ++ch) {
    const int buf = ch & 1;
    if (ch + 1 < NCHUNK) {
      stage(buf ^ 1, ch + 1);
      // allow the just-issued slab to stay in flight; wait for slab `ch`
      if (wave == 0) __builtin_amdgcn_s_wait_tensorcnt(2);
      else           __builtin_amdgcn_s_wait_tensorcnt(1);
    } else {
      __builtin_amdgcn_s_wait_tensorcnt(0);
    }
    __syncthreads();   // slab `ch` visible to all waves

    const _Float16* la = &sA[buf][(lane & 15) * KC];
    const _Float16* lb = &sB[buf][wave][(lane & 15) * KC];
#pragma unroll
    for (int kk = 0; kk < KC; kk += 32) {
      v16h a = frag_a_ptr(la + kk, khalf);
      v16h b = frag_b_ptr(lb + kk, khalf);
      c = __builtin_amdgcn_wmma_f32_16x16x32_f16(false, a, false, b,
                                                 (short)0, c, false, false);
    }
    __syncthreads();   // done reading `buf` before it is re-staged
  }

  const int n1 = n0 + (lane & 15);
  const int mh = khalf * 8;
#pragma unroll
  for (int r = 0; r < 8; ++r) {
    const int b = bm + mh + r;
    const size_t idx = (size_t)b * N1_DIM + n1;
    const float vv = v[idx];
    const float mm = mask[idx];
    out[idx] = vv + ALPHA * (c[r] / den[b] - vv) * mm;
  }
}

extern "C" void kernel_launch(void* const* d_in, const int* in_sizes, int n_in,
                              void* d_out, int out_size, void* d_ws, size_t ws_size,
                              hipStream_t stream) {
  const float* v    = (const float*)d_in[0];   // [B, N1]
  const float* mask = (const float*)d_in[1];   // [B, N1]
  const float* kern = (const float*)d_in[2];   // [N2, N1]
  float* out = (float*)d_out;

  char* ws = (char*)d_ws;
  _Float16* vh  = (_Float16*)(ws);                        // 512 KB
  _Float16* kh  = (_Float16*)(ws + (512ull << 10));       // 512 KB
  _Float16* kth = (_Float16*)(ws + (1024ull << 10));      // 512 KB
  _Float16* wh  = (_Float16*)(ws + (1536ull << 10));      // 2 MB
  float*    kw  = (float*)   (ws + (3584ull << 10));      // 4 KB
  float*    den = (float*)   (ws + (3584ull << 10) + 4096);

  prep_v_kernel<<<(B_DIM * N1_DIM) / 256, 256, 0, stream>>>(v, vh);
  prep_k_kernel<<<N2_DIM, N1_DIM, 0, stream>>>(kern, kh, kth, kw);
  gemm1_w_kernel<<<(B_DIM / 16) * (N2_DIM / 16) / 8, 256, 0, stream>>>(vh, kh, kw, wh);
  den_kernel<<<B_DIM, 256, 0, stream>>>(wh, den);
  gemm2_out_kernel<<<(B_DIM / 16) * (N1_DIM / 16 / 8), 256, 0, stream>>>(
      wh, kth, den, v, mask, out);
}
